// TwoStageBrainTumorDetector_53180285059653
// MI455X (gfx1250) — matrix operand order repaired and checked by reading, not compiled, not run on hardware
//
#include <hip/hip_runtime.h>
#include <math.h>

// ---------------------------------------------------------------------------
// MI455X (gfx1250) implementation. Every conv / FC is an implicit GEMM on
// v_wmma_f32_16x16x32_f16 (wave32, 16x16 tiles, K-steps of 32, f32 accum).
//
// - Activations stored f16: halves HBM traffic (23.3 TB/s roofline) and feeds
//   WMMA directly. Conv weights pre-packed per launch into ISA 7.12.2
//   A-fragment order; FC weights pre-packed into B-fragment order -> hot
//   loops use contiguous 32B fragment loads only.
// - Conv kernels templated on (Cin,H,W,KHW): all index math is compile-time
//   constant division; halo handling is branch-free (clamped address +
//   cndmask zero), so loads clause up instead of one s_wait per element.
// - Workgroups of up to 8 waves (one Cout-tile each) cooperatively stage the
//   shared 32x16 im2col B fragment into LDS (double buffered, 1 barrier per
//   k-step); each wave then reads it as one contiguous 32B ds load.
// - The 411MB classifier conv2 output is never materialized: its 32x32
//   avg-pool is fused into the conv epilogue via atomic accumulation.
// ---------------------------------------------------------------------------

typedef __attribute__((ext_vector_type(16))) _Float16 v16h;
typedef __attribute__((ext_vector_type(8)))  float    v8f;

union AFrag { v16h v; _Float16 e[16]; };
union CAcc  { v8f  v; float     e[8]; };

#define BATCH 32
#define NANCH 3136           // 28*28*4
#define SCORE_T 0.7f
#define IOU_T 0.3f

// ------------------------- weight fragment packing -------------------------
// A-matrix (16x32 f16) VGPR layout per ISA 7.12.2:
//   lane m in [0,15]  holds row M=m,    halves h: K = (h&7) + (h>>3)*16   (g=0)
//   lane m in [16,31] holds row M=m-16, halves h: K = 8 + (h&7)+(h>>3)*16 (g=1)
__global__ void prep_wfrag_kernel(_Float16* __restrict__ dst,
                                  const float* __restrict__ w,
                                  int Cout, int Cin, int KHW)
{
    const int KK = KHW * KHW;
    const int K  = Cin * KK;
    const int Ks = (K + 31) >> 5;
    const int Mt = (Cout + 15) >> 4;
    const size_t total = (size_t)Mt * Ks * 512;
    for (size_t t = blockIdx.x * (size_t)blockDim.x + threadIdx.x;
         t < total; t += gridDim.x * (size_t)blockDim.x) {
        int h    = (int)(t & 15);
        int lane = (int)((t >> 4) & 31);
        size_t q = t >> 9;
        int ks   = (int)(q % Ks);
        int mt   = (int)(q / Ks);
        int g = lane >> 4;
        int m = lane & 15;
        int k = (ks << 5) + (g << 3) + (h & 7) + ((h >> 3) << 4);
        int co = mt * 16 + m;
        float v = 0.f;
        if (co < Cout && k < K) v = w[(size_t)co * K + k];
        dst[t] = (_Float16)v;
    }
}

// B-matrix (32x16 f16) fragment layout: column = lane&15, K split across lane
// halves: K = ks*32 + (lane>>4)*8 + (h&7) + (h>>3)*16.  Source w is row-major [K,N].
__global__ void prep_bfrag_kernel(_Float16* __restrict__ dst,
                                  const float* __restrict__ w, int K, int N)
{
    const int Ks = K >> 5;               // K % 32 == 0 for all FC layers
    const int Nt = (N + 15) >> 4;
    const size_t total = (size_t)Nt * Ks * 512;
    for (size_t t = blockIdx.x * (size_t)blockDim.x + threadIdx.x;
         t < total; t += gridDim.x * (size_t)blockDim.x) {
        int h    = (int)(t & 15);
        int lane = (int)((t >> 4) & 31);
        size_t q = t >> 9;
        int ks   = (int)(q % Ks);
        int nt   = (int)(q / Ks);
        int k = (ks << 5) + ((lane >> 4) << 3) + (h & 7) + ((h >> 3) << 4);
        int n = nt * 16 + (lane & 15);
        float v = (n < N) ? w[(size_t)k * N + n] : 0.f;
        dst[t] = (_Float16)v;
    }
}

// fold conv-bias + eval-mode BN (mean=0, var=1):  y = conv*s + (cb*s + bb)
__global__ void fold_bn_kernel(float* __restrict__ scale, float* __restrict__ bias,
                               const float* __restrict__ convb,
                               const float* __restrict__ g,
                               const float* __restrict__ bb,
                               int C, int has_bn)
{
    int i = threadIdx.x;
    if (i < C) {
        float s  = has_bn ? g[i] * rsqrtf(1.0f + 1e-5f) : 1.0f;
        float b2 = has_bn ? bb[i] : 0.0f;
        scale[i] = s;
        bias[i]  = convb[i] * s + b2;
    }
}

__global__ void f32_to_f16_kernel(const float* __restrict__ in,
                                  _Float16* __restrict__ out, size_t n)
{
    size_t t = blockIdx.x * (size_t)blockDim.x + threadIdx.x;
    if (t < n) out[t] = (_Float16)in[t];
}

// ------------------------------ WMMA conv ----------------------------------
// Block = (32, WAVES): WAVES waves share one 16-pixel tile, each owns one
// 16-Cout tile. B fragment staged to LDS cooperatively, double-buffered.
template<int CIN, int H, int W, int KHW, int WAVES>
__global__ void __launch_bounds__(32 * WAVES)
conv_wmma_t(const _Float16* __restrict__ in,
            const _Float16* __restrict__ wfrag,
            const float* __restrict__ scale,
            const float* __restrict__ bias,
            _Float16* __restrict__ out16,
            float* __restrict__ out32,
            float* __restrict__ pool49,   // fused 32x32 avg-pool accum (or null)
            int Cout, int relu)
{
    constexpr int KK  = KHW * KHW;
    constexpr int K   = CIN * KK;
    constexpr int Ks  = (K + 31) / 32;
    constexpr int HW  = H * W;
    constexpr int pad = KHW >> 1;
    constexpr int E   = 512 / (32 * WAVES);   // staged elements per thread
    static_assert(512 % (32 * WAVES) == 0, "staging split");
    static_assert((BATCH * HW) % 16 == 0, "pixel tiles exact");

    __shared__ _Float16 lsB[2][512];

    const int lane = threadIdx.x;
    const int wv   = threadIdx.y;
    const int tid  = wv * 32 + lane;
    const int g    = lane >> 4;
    const int lcol = lane & 15;

    // output pixel of this lane's C column (constant divides -> magic muls)
    const int n   = blockIdx.x * 16 + lcol;
    const int b   = n / HW;
    const int rem = n - b * HW;
    const int py  = rem / W;
    const int px  = rem - py * W;

    const int mt = blockIdx.y * WAVES + wv;
    const _Float16* wf = wfrag + (size_t)mt * Ks * 512;

    CAcc acc = {};
    for (int ks = 0; ks < Ks; ++ks) {
        const int p = ks & 1;
        // ---- cooperative, branch-free stage of B fragment into LDS ----
        #pragma unroll
        for (int e = 0; e < E; ++e) {
            int idx = tid * E + e;            // fragment order: ln*16 + h
            int ln  = idx >> 4;
            int h   = idx & 15;
            int k   = (ks << 5) + ((ln >> 4) << 3) + (h & 7) + ((h >> 3) << 4);
            int col = ln & 15;
            int ns  = blockIdx.x * 16 + col;
            int bs  = ns / HW;
            int rs  = ns - bs * HW;
            int ys  = rs / W;
            int xs  = rs - ys * W;
            // clamp indices -> unconditional load, then cndmask-select zero
            int kc  = ((K & 31) == 0) ? k : (k < K ? k : K - 1);
            int cin = kc / KK;
            int r   = kc - cin * KK;
            int kh  = r / KHW, kw = r - kh * KHW;
            int iy  = ys + kh - pad, ix = xs + kw - pad;
            bool inb = ((unsigned)iy < (unsigned)H) &
                       ((unsigned)ix < (unsigned)W) &
                       (((K & 31) == 0) | (k < K));
            int iyc = iy < 0 ? 0 : (iy >= H ? H - 1 : iy);
            int ixc = ix < 0 ? 0 : (ix >= W ? W - 1 : ix);
            _Float16 v = in[((size_t)bs * CIN + cin) * HW + iyc * W + ixc];
            lsB[p][idx] = inb ? v : (_Float16)0.f;
        }
        __syncthreads();

        AFrag a, bf;
        a.v = *reinterpret_cast<const v16h*>(wf + (size_t)ks * 512 + lane * 16);
        if (ks + 1 < Ks)  // -> global_prefetch of next weight fragment
            __builtin_prefetch(wf + (size_t)(ks + 1) * 512 + lane * 16, 0, 0);
        bf.v = *reinterpret_cast<const v16h*>(&lsB[p][lane * 16]);

        acc.v = __builtin_amdgcn_wmma_f32_16x16x32_f16(
                    false, a.v, false, bf.v, (short)0, acc.v, false, false);
    }

    // C layout: VGPR vg -> M = vg + 8*(lane>=16), N = lane&15
    #pragma unroll
    for (int vg = 0; vg < 8; ++vg) {
        int m  = vg + (g << 3);
        int co = mt * 16 + m;
        if (co < Cout) {
            float v = acc.e[vg] * scale[co] + bias[co];
            if (relu) v = fmaxf(v, 0.f);
            size_t oidx = ((size_t)b * Cout + co) * HW + rem;
            if (out16) out16[oidx] = (_Float16)v;
            if (out32) out32[oidx] = v;
            if (pool49) {
                int pb = (py >> 5) * 7 + (px >> 5);     // 224/32 = 7
                atomicAdd(&pool49[((size_t)b * Cout + co) * 49 + pb],
                          v * (1.0f / 1024.0f));
            }
        }
    }
}

// ------------------------------ WMMA GEMM ----------------------------------
// C[M,N] = relu(A[M,K]*B + bias[N]).  A is f32 row-major (K % 32 == 0, rows
// 32B aligned -> 4x b128 loads per step); B pre-packed f16 fragments.
__global__ void __launch_bounds__(32)
gemm_wmma_kernel(const float* __restrict__ A, const _Float16* __restrict__ Bfrag,
                 const float* __restrict__ bias, float* __restrict__ C,
                 int M, int K, int N, int relu)
{
    const int lane = threadIdx.x;
    const int g    = lane >> 4;
    const int lidx = lane & 15;
    const int ncol = blockIdx.x * 16 + lidx;
    const int mrow = blockIdx.y * 16 + lidx;
    const int Ks   = K >> 5;

    const int mc = (mrow < M) ? mrow : (M - 1);   // clamp: no hot-loop guards
    const float* Arow = A + (size_t)mc * K;
    const _Float16* bfp = Bfrag + (size_t)blockIdx.x * Ks * 512;

    CAcc acc = {};
    for (int ks = 0; ks < Ks; ++ks) {
        int ka0 = (ks << 5) + (g << 3);
        float4 p0 = *reinterpret_cast<const float4*>(Arow + ka0);
        float4 p1 = *reinterpret_cast<const float4*>(Arow + ka0 + 4);
        float4 p2 = *reinterpret_cast<const float4*>(Arow + ka0 + 16);
        float4 p3 = *reinterpret_cast<const float4*>(Arow + ka0 + 20);
        AFrag a, b;
        a.e[0]  = (_Float16)p0.x; a.e[1]  = (_Float16)p0.y;
        a.e[2]  = (_Float16)p0.z; a.e[3]  = (_Float16)p0.w;
        a.e[4]  = (_Float16)p1.x; a.e[5]  = (_Float16)p1.y;
        a.e[6]  = (_Float16)p1.z; a.e[7]  = (_Float16)p1.w;
        a.e[8]  = (_Float16)p2.x; a.e[9]  = (_Float16)p2.y;
        a.e[10] = (_Float16)p2.z; a.e[11] = (_Float16)p2.w;
        a.e[12] = (_Float16)p3.x; a.e[13] = (_Float16)p3.y;
        a.e[14] = (_Float16)p3.z; a.e[15] = (_Float16)p3.w;
        b.v = *reinterpret_cast<const v16h*>(bfp + (size_t)ks * 512 + lane * 16);
        if (ks + 1 < Ks)
            __builtin_prefetch(bfp + (size_t)(ks + 1) * 512 + lane * 16, 0, 0);
        acc.v = __builtin_amdgcn_wmma_f32_16x16x32_f16(
                    false, a.v, false, b.v, (short)0, acc.v, false, false);
    }
    #pragma unroll
    for (int vg = 0; vg < 8; ++vg) {
        int m = blockIdx.y * 16 + vg + (g << 3);
        if (m < M && ncol < N) {
            float v = acc.e[vg] + bias[ncol];
            if (relu) v = fmaxf(v, 0.f);
            C[(size_t)m * N + ncol] = v;
        }
    }
}

// ------------------------------ small ops ----------------------------------
__global__ void maxpool_kernel(const _Float16* __restrict__ in,
                               _Float16* __restrict__ out, int BC, int H, int W)
{
    int Ho = H >> 1, Wo = W >> 1;
    size_t total = (size_t)BC * Ho * Wo;
    size_t t = blockIdx.x * (size_t)blockDim.x + threadIdx.x;
    if (t >= total) return;
    int wo = (int)(t % Wo);
    size_t q = t / Wo;
    int ho = (int)(q % Ho);
    int bc = (int)(q / Ho);
    const _Float16* p = in + ((size_t)bc * H + ho * 2) * W + wo * 2;
    float a = (float)p[0], b = (float)p[1], c = (float)p[W], d = (float)p[W + 1];
    out[t] = (_Float16)fmaxf(fmaxf(a, b), fmaxf(c, d));
}

__global__ void decode_kernel(const float* __restrict__ cls,
                              const float* __restrict__ reg,
                              float* __restrict__ probs,
                              float* __restrict__ boxes)
{
    int t = blockIdx.x * blockDim.x + threadIdx.x;
    if (t >= BATCH * NANCH) return;
    int b = t / NANCH, i = t - b * NANCH;
    int a = i & 3;
    int cell = i >> 2;
    int y = cell / 28, x = cell - y * 28;
    const int HW = 784;
    size_t cbase = (size_t)b * 8 * HW + y * 28 + x;
    float c0 = cls[cbase + (size_t)(a * 2 + 0) * HW];
    float c1 = cls[cbase + (size_t)(a * 2 + 1) * HW];
    probs[t] = 1.0f / (1.0f + expf(c0 - c1));       // softmax [...,1]
    size_t rbase = (size_t)b * 16 * HW + y * 28 + x;
    float o0 = reg[rbase + (size_t)(a * 4 + 0) * HW];
    float o1 = reg[rbase + (size_t)(a * 4 + 1) * HW];
    float o2 = reg[rbase + (size_t)(a * 4 + 2) * HW];
    float o3 = reg[rbase + (size_t)(a * 4 + 3) * HW];
    float cx = (x + 0.5f) / 28.f + o0 * 0.1f;
    float cy = (y + 0.5f) / 28.f + o1 * 0.1f;
    const float sz0 = 16.f / 224.f;
    float s = sz0 * (float)(1 << a);                // 16,32,64,128 / 224
    float w = s * expf(o2 * 0.2f);
    float h = s * expf(o3 * 0.2f);
    boxes[(size_t)t * 4 + 0] = fminf(fmaxf(cx - w * 0.5f, 0.f), 1.f);
    boxes[(size_t)t * 4 + 1] = fminf(fmaxf(cy - h * 0.5f, 0.f), 1.f);
    boxes[(size_t)t * 4 + 2] = fminf(fmaxf(cx + w * 0.5f, 0.f), 1.f);
    boxes[(size_t)t * 4 + 3] = fminf(fmaxf(cy + h * 0.5f, 0.f), 1.f);
}

// Greedy NMS, 10 fixed iterations, one block per image. LDS-resident scores.
__global__ void __launch_bounds__(256)
nms_kernel(const float* __restrict__ probs, const float* __restrict__ boxes,
           float* __restrict__ outBoxes, float* __restrict__ outScores,
           float* __restrict__ outOks, float* __restrict__ hasprop)
{
    __shared__ float s[NANCH];
    __shared__ float area[NANCH];
    __shared__ float rval[256];
    __shared__ int   ridx[256];
    __shared__ float bbox[4];
    __shared__ int   sh_bi, sh_ok;

    int b = blockIdx.x, tid = threadIdx.x;
    const float* pb = probs + (size_t)b * NANCH;
    const float* bx = boxes + (size_t)b * NANCH * 4;

    for (int i = tid; i < NANCH; i += 256) {
        float p = pb[i];
        s[i]    = (p > SCORE_T) ? p : -__builtin_inff();
        area[i] = (bx[i * 4 + 2] - bx[i * 4 + 0]) * (bx[i * 4 + 3] - bx[i * 4 + 1]);
    }
    __syncthreads();

    for (int it = 0; it < 10; ++it) {
        // block argmax with first-index tie-breaking (matches jnp.argmax)
        float bvl = -__builtin_inff();
        int   bix = NANCH;
        for (int i = tid; i < NANCH; i += 256) {
            float v = s[i];
            if (v > bvl || (v == bvl && i < bix)) { bvl = v; bix = i; }
        }
        rval[tid] = bvl; ridx[tid] = bix;
        __syncthreads();
        for (int off = 128; off > 0; off >>= 1) {
            if (tid < off) {
                float v2 = rval[tid + off]; int i2 = ridx[tid + off];
                if (v2 > rval[tid] || (v2 == rval[tid] && i2 < ridx[tid])) {
                    rval[tid] = v2; ridx[tid] = i2;
                }
            }
            __syncthreads();
        }
        if (tid == 0) {
            int i = (ridx[0] >= NANCH) ? 0 : ridx[0];
            int ok = (rval[0] > -__builtin_inff()) ? 1 : 0;
            sh_bi = i; sh_ok = ok;
            bbox[0] = bx[i * 4 + 0]; bbox[1] = bx[i * 4 + 1];
            bbox[2] = bx[i * 4 + 2]; bbox[3] = bx[i * 4 + 3];
            float okf = ok ? 1.f : 0.f;
            int o = b * 10 + it;
            outBoxes[o * 4 + 0] = bbox[0] * okf;
            outBoxes[o * 4 + 1] = bbox[1] * okf;
            outBoxes[o * 4 + 2] = bbox[2] * okf;
            outBoxes[o * 4 + 3] = bbox[3] * okf;
            outScores[o] = pb[i] * okf;
            outOks[o]    = okf;
            if (it == 0) hasprop[b] = okf;   // ok0 <=> any(prob > T)
        }
        __syncthreads();
        int ii = sh_bi, ok = sh_ok;
        float ai = area[ii];
        float b0 = bbox[0], b1 = bbox[1], b2 = bbox[2], b3 = bbox[3];
        for (int j = tid; j < NANCH; j += 256) {
            float xx1 = fmaxf(b0, bx[j * 4 + 0]);
            float yy1 = fmaxf(b1, bx[j * 4 + 1]);
            float xx2 = fminf(b2, bx[j * 4 + 2]);
            float yy2 = fminf(b3, bx[j * 4 + 3]);
            float inter = fmaxf(xx2 - xx1, 0.f) * fmaxf(yy2 - yy1, 0.f);
            float iou = inter / (ai + area[j] - inter);
            if (ok && (iou > IOU_T || j == ii)) s[j] = -__builtin_inff();
        }
        __syncthreads();
    }
}

__global__ void pack_final_kernel(const float* __restrict__ logits,
                                  const float* __restrict__ hasprop,
                                  float* __restrict__ final_out)
{
    int t = threadIdx.x;
    if (t < 64) {
        int b = t >> 1, j = t & 1;
        float bg = (j == 0) ? 1.0f : 0.0f;
        final_out[t] = (hasprop[b] > 0.5f) ? logits[b * 2 + j] : bg;
    }
}

// ------------------------------ orchestration ------------------------------
static inline size_t frag_elems(int Cout, int Cin, int KHW) {
    int K = Cin * KHW * KHW;
    return (size_t)((Cout + 15) / 16) * ((K + 31) / 32) * 512;
}
static inline size_t bfrag_elems(int K, int N) {
    return (size_t)((N + 15) / 16) * (K / 32) * 512;
}

extern "C" void kernel_launch(void* const* d_in, const int* in_sizes, int n_in,
                              void* d_out, int out_size, void* d_ws, size_t ws_size,
                              hipStream_t stream)
{
    const float* x    = (const float*)d_in[0];
    const float* r1w  = (const float*)d_in[1];  const float* r1b = (const float*)d_in[2];
    const float* g1   = (const float*)d_in[3];  const float* b1  = (const float*)d_in[4];
    const float* r2w  = (const float*)d_in[5];  const float* r2b = (const float*)d_in[6];
    const float* g2   = (const float*)d_in[7];  const float* b2  = (const float*)d_in[8];
    const float* r3w  = (const float*)d_in[9];  const float* r3b = (const float*)d_in[10];
    const float* g3   = (const float*)d_in[11]; const float* b3  = (const float*)d_in[12];
    const float* rpw  = (const float*)d_in[13]; const float* rpb = (const float*)d_in[14];
    const float* clsw = (const float*)d_in[15]; const float* clsb= (const float*)d_in[16];
    const float* regw = (const float*)d_in[17]; const float* regb= (const float*)d_in[18];
    const float* c1w  = (const float*)d_in[19]; const float* c1b = (const float*)d_in[20];
    const float* cg1  = (const float*)d_in[21]; const float* cb1 = (const float*)d_in[22];
    const float* c2w  = (const float*)d_in[23]; const float* c2b = (const float*)d_in[24];
    const float* cg2  = (const float*)d_in[25]; const float* cb2 = (const float*)d_in[26];
    const float* fc1w = (const float*)d_in[27]; const float* fc1b= (const float*)d_in[28];
    const float* fc2w = (const float*)d_in[29]; const float* fc2b= (const float*)d_in[30];
    const float* fc3w = (const float*)d_in[31]; const float* fc3b= (const float*)d_in[32];
    const float* fc4w = (const float*)d_in[33]; const float* fc4b= (const float*)d_in[34];

    char* base = (char*)d_ws;
    size_t off = 0;
    auto alloc = [&](size_t bytes) -> void* {
        void* p = base + off;
        off = (off + bytes + 255) & ~(size_t)255;
        return p;
    };

    // activations (f16) --------------------------------------------------
    _Float16* x16   = (_Float16*)alloc((size_t)BATCH*4*224*224*2);
    _Float16* h1pre = (_Float16*)alloc((size_t)BATCH*32*224*224*2);
    _Float16* h1    = (_Float16*)alloc((size_t)BATCH*32*112*112*2);
    _Float16* h2pre = (_Float16*)alloc((size_t)BATCH*64*112*112*2);
    _Float16* h2    = (_Float16*)alloc((size_t)BATCH*64*56*56*2);
    _Float16* h3pre = (_Float16*)alloc((size_t)BATCH*128*56*56*2);
    _Float16* h3    = (_Float16*)alloc((size_t)BATCH*128*28*28*2);
    _Float16* rp    = (_Float16*)alloc((size_t)BATCH*256*28*28*2);
    float* clsbuf   = (float*)alloc((size_t)BATCH*8*784*4);
    float* regbuf   = (float*)alloc((size_t)BATCH*16*784*4);
    float* probs    = (float*)alloc((size_t)BATCH*NANCH*4);
    float* boxes    = (float*)alloc((size_t)BATCH*NANCH*4*4);
    float* hasprop  = (float*)alloc(BATCH*4);
    _Float16* f1    = (_Float16*)alloc((size_t)BATCH*64*224*224*2);
    float* pooled   = (float*)alloc((size_t)BATCH*6272*4);
    float* a1       = (float*)alloc((size_t)BATCH*512*4);
    float* a2       = (float*)alloc((size_t)BATCH*256*4);
    float* a3       = (float*)alloc((size_t)BATCH*128*4);
    float* logits   = (float*)alloc((size_t)BATCH*2*4);

    // packed weight fragments + folded BN params -------------------------
    _Float16* wf_r1  = (_Float16*)alloc(frag_elems(32, 4, 3)   * 2);
    _Float16* wf_r2  = (_Float16*)alloc(frag_elems(64, 32, 3)  * 2);
    _Float16* wf_r3  = (_Float16*)alloc(frag_elems(128, 64, 3) * 2);
    _Float16* wf_rp  = (_Float16*)alloc(frag_elems(256, 128, 3)* 2);
    _Float16* wf_cls = (_Float16*)alloc(frag_elems(8, 256, 1)  * 2);
    _Float16* wf_reg = (_Float16*)alloc(frag_elems(16, 256, 1) * 2);
    _Float16* wf_c1  = (_Float16*)alloc(frag_elems(64, 4, 3)   * 2);
    _Float16* wf_c2  = (_Float16*)alloc(frag_elems(128, 64, 3) * 2);
    _Float16* bf_fc1 = (_Float16*)alloc(bfrag_elems(6272, 512) * 2);
    _Float16* bf_fc2 = (_Float16*)alloc(bfrag_elems(512, 256)  * 2);
    _Float16* bf_fc3 = (_Float16*)alloc(bfrag_elems(256, 128)  * 2);
    _Float16* bf_fc4 = (_Float16*)alloc(bfrag_elems(128, 2)    * 2);
    float* sb = (float*)alloc(16 * 256 * 4);   // 8 convs x (scale, bias) x 256
    float* sc_r1 = sb+0*256,  *bi_r1 = sb+1*256;
    float* sc_r2 = sb+2*256,  *bi_r2 = sb+3*256;
    float* sc_r3 = sb+4*256,  *bi_r3 = sb+5*256;
    float* sc_rp = sb+6*256,  *bi_rp = sb+7*256;
    float* sc_cl = sb+8*256,  *bi_cl = sb+9*256;
    float* sc_rg = sb+10*256, *bi_rg = sb+11*256;
    float* sc_c1 = sb+12*256, *bi_c1 = sb+13*256;
    float* sc_c2 = sb+14*256, *bi_c2 = sb+15*256;

    if (off > ws_size) return;   // workspace too small; nothing safe to do

    float* out       = (float*)d_out;
    float* outFinal  = out;            // 32*2    = 64
    float* outBoxes  = out + 64;       // 32*10*4 = 1280
    float* outScores = out + 1344;     // 32*10   = 320
    float* outOks    = out + 1664;     // 32*10   = 320

    auto prep = [&](_Float16* dst, const float* w, int Cout, int Cin, int KHW) {
        size_t total = frag_elems(Cout, Cin, KHW);
        int blocks = (int)((total + 255) / 256);
        prep_wfrag_kernel<<<blocks, 256, 0, stream>>>(dst, w, Cout, Cin, KHW);
    };
    auto prepB = [&](_Float16* dst, const float* w, int K, int N) {
        size_t total = bfrag_elems(K, N);
        int blocks = (int)((total + 255) / 256);
        prep_bfrag_kernel<<<blocks, 256, 0, stream>>>(dst, w, K, N);
    };
    auto fold = [&](float* sc, float* bi, const float* cb, const float* g,
                    const float* bb, int C, int hasbn) {
        fold_bn_kernel<<<1, 256, 0, stream>>>(sc, bi, cb, g, bb, C, hasbn);
    };
    auto pool2 = [&](const _Float16* in, _Float16* o, int BC, int H, int W) {
        size_t total = (size_t)BC * (H / 2) * (W / 2);
        maxpool_kernel<<<(int)((total + 255) / 256), 256, 0, stream>>>(in, o, BC, H, W);
    };
    auto gemm = [&](const float* A, const _Float16* Bf, const float* bias, float* C,
                    int M, int K, int N, int relu) {
        dim3 grid((N + 15) / 16, (M + 15) / 16);
        gemm_wmma_kernel<<<grid, 32, 0, stream>>>(A, Bf, bias, C, M, K, N, relu);
    };

    // ---- prep: pack weights + fold BN ----
    prep(wf_r1,  r1w, 32, 4, 3);     fold(sc_r1, bi_r1, r1b, g1, b1, 32, 1);
    prep(wf_r2,  r2w, 64, 32, 3);    fold(sc_r2, bi_r2, r2b, g2, b2, 64, 1);
    prep(wf_r3,  r3w, 128, 64, 3);   fold(sc_r3, bi_r3, r3b, g3, b3, 128, 1);
    prep(wf_rp,  rpw, 256, 128, 3);  fold(sc_rp, bi_rp, rpb, nullptr, nullptr, 256, 0);
    prep(wf_cls, clsw, 8, 256, 1);   fold(sc_cl, bi_cl, clsb, nullptr, nullptr, 8, 0);
    prep(wf_reg, regw, 16, 256, 1);  fold(sc_rg, bi_rg, regb, nullptr, nullptr, 16, 0);
    prep(wf_c1,  c1w, 64, 4, 3);     fold(sc_c1, bi_c1, c1b, cg1, cb1, 64, 1);
    prep(wf_c2,  c2w, 128, 64, 3);   fold(sc_c2, bi_c2, c2b, cg2, cb2, 128, 1);
    prepB(bf_fc1, fc1w, 6272, 512);
    prepB(bf_fc2, fc2w, 512, 256);
    prepB(bf_fc3, fc3w, 256, 128);
    prepB(bf_fc4, fc4w, 128, 2);

    // ---- input to f16 ----
    {
        size_t n = (size_t)BATCH * 4 * 224 * 224;
        f32_to_f16_kernel<<<(int)((n + 255) / 256), 256, 0, stream>>>(x, x16, n);
    }

    // ---- stage 1: RPN (templated WMMA convs) ----
    {   // r1: 4->32 @224, BN+ReLU.  Mt=2, WAVES=2
        dim3 grid(BATCH*224*224/16, 1);
        conv_wmma_t<4,224,224,3,2><<<grid, dim3(32,2), 0, stream>>>(
            x16, wf_r1, sc_r1, bi_r1, h1pre, nullptr, nullptr, 32, 1);
    }
    pool2(h1pre, h1, BATCH * 32, 224, 224);
    {   // r2: 32->64 @112.  Mt=4, WAVES=4
        dim3 grid(BATCH*112*112/16, 1);
        conv_wmma_t<32,112,112,3,4><<<grid, dim3(32,4), 0, stream>>>(
            h1, wf_r2, sc_r2, bi_r2, h2pre, nullptr, nullptr, 64, 1);
    }
    pool2(h2pre, h2, BATCH * 64, 112, 112);
    {   // r3: 64->128 @56.  Mt=8, WAVES=8
        dim3 grid(BATCH*56*56/16, 1);
        conv_wmma_t<64,56,56,3,8><<<grid, dim3(32,8), 0, stream>>>(
            h2, wf_r3, sc_r3, bi_r3, h3pre, nullptr, nullptr, 128, 1);
    }
    pool2(h3pre, h3, BATCH * 128, 56, 56);
    {   // rp: 128->256 @28, bias+ReLU.  Mt=16, WAVES=8, grid.y=2
        dim3 grid(BATCH*28*28/16, 2);
        conv_wmma_t<128,28,28,3,8><<<grid, dim3(32,8), 0, stream>>>(
            h3, wf_rp, sc_rp, bi_rp, rp, nullptr, nullptr, 256, 1);
    }
    {   // cls head: 256->8 1x1, f32 out
        dim3 grid(BATCH*28*28/16, 1);
        conv_wmma_t<256,28,28,1,1><<<grid, dim3(32,1), 0, stream>>>(
            rp, wf_cls, sc_cl, bi_cl, nullptr, clsbuf, nullptr, 8, 0);
    }
    {   // reg head: 256->16 1x1, f32 out
        dim3 grid(BATCH*28*28/16, 1);
        conv_wmma_t<256,28,28,1,1><<<grid, dim3(32,1), 0, stream>>>(
            rp, wf_reg, sc_rg, bi_rg, nullptr, regbuf, nullptr, 16, 0);
    }

    decode_kernel<<<(BATCH * NANCH + 255) / 256, 256, 0, stream>>>(clsbuf, regbuf,
                                                                   probs, boxes);
    nms_kernel<<<BATCH, 256, 0, stream>>>(probs, boxes, outBoxes, outScores,
                                          outOks, hasprop);

    // ---- stage 2: classifier (avg-pool fused into conv2 epilogue) ----
    {   // c1: 4->64 @224, BN+ReLU.  Mt=4, WAVES=4
        dim3 grid(BATCH*224*224/16, 1);
        conv_wmma_t<4,224,224,3,4><<<grid, dim3(32,4), 0, stream>>>(
            x16, wf_c1, sc_c1, bi_c1, f1, nullptr, nullptr, 64, 1);
    }
    hipMemsetAsync(pooled, 0, (size_t)BATCH * 6272 * sizeof(float), stream);
    {   // c2: 64->128 @224, BN+ReLU, fused 32x32 avg-pool.  Mt=8, WAVES=8
        dim3 grid(BATCH*224*224/16, 1);
        conv_wmma_t<64,224,224,3,8><<<grid, dim3(32,8), 0, stream>>>(
            f1, wf_c2, sc_c2, bi_c2, nullptr, nullptr, pooled, 128, 1);
    }

    gemm(pooled, bf_fc1, fc1b, a1, BATCH, 6272, 512, 1);
    gemm(a1, bf_fc2, fc2b, a2, BATCH, 512, 256, 1);
    gemm(a2, bf_fc3, fc3b, a3, BATCH, 256, 128, 1);
    gemm(a3, bf_fc4, fc4b, logits, BATCH, 128, 2, 0);

    pack_final_kernel<<<1, 64, 0, stream>>>(logits, hasprop, outFinal);
}
